// tsp_coder_simply_56495999811908
// MI455X (gfx1250) — matrix operand order repaired and checked by reading, not compiled, not run on hardware
//
#include <hip/hip_runtime.h>

// ---------------- CDNA5 (gfx1250) fused TSP-coder forward ----------------
// All 1x1 convs -> wave-tiled WMMA GEMMs (v_wmma_f32_16x16x32_f16, fp32 acc)
// with LDS-staged activation tiles (fp32 -> f16 convert on stage).
// Gathers / softmax / einsum-agg -> fused VALU kernels, L2-resident buffers.

typedef __attribute__((ext_vector_type(16))) _Float16 v16h;
typedef __attribute__((ext_vector_type(8)))  _Float16 v8h;
typedef __attribute__((ext_vector_type(8)))  float    v8f;

#define BB 4
#define NNODE 1024
#define KNN 16
#define TT 16
#define PN 4096    // B*N
#define PE 65536   // B*N*K

// LDS X-tile layout: [64 cols][48 halves] (only 32 K used; stride 96B keeps
// every 8-half chunk 16B-aligned and spreads banks).
#define XSTRIDE 48

// ---------------------------------------------------------------------------
// Generic per-position linear:  Y[M][P] = epilogue( W[M][KC] * X[KC][P] )
// One wave per 16x16 tile; 4 waves (128 thr) per block cover a 64-col tile.
// Block cooperatively stages X[k0..k0+31][col0..col0+63] -> f16 in LDS,
// transposed to [col][k] so each lane reads 2 contiguous 16B chunks.
// Epilogue: +bias, optional BN affine (g,bt), optional ReLU, optional +res.
// ---------------------------------------------------------------------------
template<int KC, bool BN, bool RELU, bool RES>
__global__ __launch_bounds__(128)
void gemm_wmma_k(const float* __restrict__ W, const float* __restrict__ bias,
                 const float* __restrict__ g, const float* __restrict__ bt,
                 const float* __restrict__ X, const float* __restrict__ res,
                 float* __restrict__ Y, int M, int P)
{
    __shared__ __align__(16) _Float16 ldsX[64 * XSTRIDE];

    const int tid  = threadIdx.x;
    const int wave = tid >> 5;
    const int lane = tid & 31;
    const int half = lane >> 4;        // wave32: lanes 0-15 / 16-31
    const int r    = lane & 15;
    const int row0 = blockIdx.y * 16;
    const int col0 = blockIdx.x * 64;
    const int col  = col0 + wave * 16 + r;
    const float* wrow = W + (size_t)(row0 + r) * KC;

    v8f acc = {};
#pragma unroll
    for (int k0 = 0; k0 < KC; k0 += 32) {
        // ---- stage X tile: 32 K-rows x 64 cols, fp32 -> f16, transposed ----
        __syncthreads();
#pragma unroll
        for (int i = 0; i < 16; ++i) {          // 128 thr * 16 = 2048 elems
            int linear = tid + 128 * i;
            int k = linear >> 6;                // 0..31
            int c = linear & 63;                // 0..63
            int kk = k0 + k;
            float v = (kk < KC) ? X[(size_t)kk * P + col0 + c] : 0.0f;
            ldsX[c * XSTRIDE + k] = (_Float16)v;
        }
        if (k0 + 32 < KC)  // gfx1250 global_prefetch_b8 for next K-tile of X
            __builtin_prefetch(X + (size_t)(k0 + 32) * P + col, 0, 0);
        __syncthreads();

        // ---- A fragment (16x32 f16): lane r = row, K = 16*half + e
        v16h a;
#pragma unroll
        for (int e = 0; e < 8; ++e) {
            int ka0 = k0 + 16 * half + e;
            int ka1 = ka0 + 8;
            a[e]     = (ka0 < KC) ? (_Float16)wrow[ka0] : (_Float16)0.0f;
            a[e + 8] = (ka1 < KC) ? (_Float16)wrow[ka1] : (_Float16)0.0f;
        }
        // ---- B fragment (32x16 f16): lane r = col, K = 16*half + e, from LDS
        const v8h* bp = (const v8h*)&ldsX[(wave * 16 + r) * XSTRIDE + 16 * half];
        v8h b0 = bp[0];
        v8h b1 = bp[1];
        v16h b;
#pragma unroll
        for (int e = 0; e < 8; ++e) { b[e] = b0[e]; b[e + 8] = b1[e]; }

        acc = __builtin_amdgcn_wmma_f32_16x16x32_f16(
                  false, a, false, b, (short)0, acc, false, false);
    }

    // C/D layout: acc[j] -> row (j + 8*half), col r of the tile.
#pragma unroll
    for (int j = 0; j < 8; ++j) {
        int m = row0 + j + 8 * half;
        float y = acc[j] + bias[m];
        if (BN)   y = y * g[m] + bt[m];
        if (RELU) y = fmaxf(y, 0.0f);
        if (RES)  y += res[(size_t)m * P + col];
        Y[(size_t)m * P + col] = y;
    }
}

// ---------------------------------------------------------------------------
// Edge-feature builders / softmax / einsum-aggregation / head (VALU kernels)
// ---------------------------------------------------------------------------
__global__ __launch_bounds__(256)
void k_build_ein(const float* __restrict__ pts, const float* __restrict__ pwt,
                 const int* __restrict__ nn,
                 float* __restrict__ ein3, float* __restrict__ eh5)
{
    int p = blockIdx.x * 256 + threadIdx.x;   // p = (b*N+n)*K + k
    int b = p >> 14;
    int n = (p >> 4) & 1023;
    int idx = nn[p];
    float cx = pts[b * 2048 + n],        cy = pts[b * 2048 + 1024 + n];
    float nx = pts[b * 2048 + idx],      ny = pts[b * 2048 + 1024 + idx];
    float pw = pwt[p];
    ein3[0 * PE + p] = nx - cx;  ein3[1 * PE + p] = ny - cy;  ein3[2 * PE + p] = pw;
    eh5[0 * PE + p] = pw;   eh5[1 * PE + p] = cx;  eh5[2 * PE + p] = cy;
    eh5[3 * PE + p] = nx;   eh5[4 * PE + p] = ny;
}

__global__ __launch_bounds__(256)
void k_build_ptsT(const float* __restrict__ pts, float* __restrict__ ptsT)
{
    int t = blockIdx.x * 256 + threadIdx.x;   // [2][PN]
    int c = t >> 12, pn = t & 4095;
    int b = pn >> 10, n = pn & 1023;
    ptsT[c * PN + pn] = pts[b * 2048 + c * 1024 + n];
}

__global__ __launch_bounds__(256)
void k_softmax16(const float* __restrict__ logits, float* __restrict__ etype)
{
    int p = blockIdx.x * 256 + threadIdx.x;
    float v[TT], mx = -3.4e38f;
#pragma unroll
    for (int t = 0; t < TT; ++t) { v[t] = logits[t * PE + p]; mx = fmaxf(mx, v[t]); }
    float s = 0.0f;
#pragma unroll
    for (int t = 0; t < TT; ++t) { v[t] = __expf(v[t] - mx); s += v[t]; }
    float inv = 1.0f / s;
#pragma unroll
    for (int t = 0; t < TT; ++t) etype[t * PE + p] = v[t] * inv;
}

// agg[f,pn] = relu(bn( selfb[f,pn] + (1/K) * sum_{t,k} etype[t,pn*K+k] *
//                       theta[t*64+f][b*N + nn[pn,k]] ))
__global__ __launch_bounds__(256)
void k_agg(const float* __restrict__ etype, const float* __restrict__ theta,
           const int* __restrict__ nn, const float* __restrict__ selfb,
           const float* __restrict__ g, const float* __restrict__ bt,
           float* __restrict__ out)
{
    int gid = blockIdx.x * 256 + threadIdx.x;   // 64*PN threads
    int f  = gid >> 12;          // /4096
    int pn = gid & 4095;
    int b  = pn >> 10;
    float acc = 0.0f;
    for (int k = 0; k < KNN; ++k) {
        int idx = nn[pn * KNN + k];
        int colt = b * NNODE + idx;
        int pe = pn * KNN + k;
#pragma unroll
        for (int t = 0; t < TT; ++t)
            acc = fmaf(etype[t * PE + pe], theta[(size_t)(t * 64 + f) * PN + colt], acc);
    }
    float y = acc * (1.0f / KNN) + selfb[f * PN + pn];
    y = y * g[f] + bt[f];
    out[f * PN + pn] = fmaxf(y, 0.0f);
}

// eh144 = concat[ w1(16) ; center n1(64) ; neighbor n1(64) ]   -> [144][PE]
__global__ __launch_bounds__(256)
void k_build_eh144(const float* __restrict__ w1, const float* __restrict__ n1,
                   const int* __restrict__ nn, float* __restrict__ eh)
{
    int c = blockIdx.y;
    int p = blockIdx.x * 256 + threadIdx.x;
    float v;
    if (c < 16)      v = w1[c * PE + p];
    else if (c < 80) v = n1[(c - 16) * PN + (p >> 4)];
    else {
        int b = p >> 14;
        v = n1[(c - 80) * PN + b * NNODE + nn[p]];
    }
    eh[(size_t)c * PE + p] = v;
}

__global__ __launch_bounds__(256)
void k_gmax(const float* __restrict__ nf, float* __restrict__ gmx)
{
    int c = threadIdx.x >> 2, b = threadIdx.x & 3;   // 256 = 64*4
    float m = -3.4e38f;
    for (int n = 0; n < NNODE; ++n)
        m = fmaxf(m, nf[c * PN + b * NNODE + n]);
    gmx[c * 4 + b] = m;
}

__global__ __launch_bounds__(256)
void k_build_cat(const float* __restrict__ nf, const float* __restrict__ gmx,
                 float* __restrict__ cat)
{
    int gid = blockIdx.x * 256 + threadIdx.x;   // [128][PN]
    int c = gid >> 12, pn = gid & 4095, b = pn >> 10;
    cat[gid] = (c < 64) ? nf[c * PN + pn] : gmx[(c - 64) * 4 + b];
}

__global__ __launch_bounds__(256)
void k_ctx2(const float* __restrict__ w, const float* __restrict__ bias,
            const float* __restrict__ h, float* __restrict__ out)
{
    int pn = blockIdx.x * 256 + threadIdx.x;
    float acc = bias[0];
#pragma unroll
    for (int c = 0; c < 64; ++c)
        acc = fmaf(w[c], h[c * PN + pn], acc);
    out[pn] = acc;   // [B,N] flat == pn
}

// ---------------------------------------------------------------------------
extern "C" void kernel_launch(void* const* d_in, const int* in_sizes, int n_in,
                              void* d_out, int out_size, void* d_ws, size_t ws_size,
                              hipStream_t stream)
{
    (void)in_sizes; (void)n_in; (void)out_size; (void)ws_size;
    const float* pts = (const float*)d_in[0];
    const float* pwt = (const float*)d_in[1];
    const int*   nn  = (const int*)d_in[2];
    auto F = [&](int i) { return (const float*)d_in[i]; };

    float* ws = (float*)d_ws;
    size_t o = 0;
    auto A = [&](size_t nel) { float* p = ws + o; o += nel; return p; };
    float* bigA  = A((size_t)144 * PE);  // time-shared: h64 / theta / eh144
    float* ein3  = A((size_t)3 * PE);
    float* eh5   = A((size_t)5 * PE);
    float* ptsT  = A((size_t)2 * PN);
    float* logit = A((size_t)16 * PE);
    float* etype = A((size_t)16 * PE);
    float* selfb = A((size_t)64 * PN);
    float* nfA   = A((size_t)64 * PN);
    float* nfB   = A((size_t)64 * PN);
    float* wfA   = A((size_t)16 * PE);
    float* wfB   = A((size_t)16 * PE);
    float* n1    = A((size_t)64 * PN);
    float* w1    = A((size_t)16 * PE);
    float* n1m   = A((size_t)64 * PN);
    float* w1m   = A((size_t)16 * PE);
    float* gmx   = A(256);
    float* cat   = A((size_t)128 * PN);
    float* hctx  = A((size_t)64 * PN);

    // ---- edge features + edge-type head ----
    k_build_ein <<<PE / 256, 256, 0, stream>>>(pts, pwt, nn, ein3, eh5);
    k_build_ptsT<<<(2 * PN) / 256, 256, 0, stream>>>(pts, ptsT);
    // et1: CBR(3->64) over PE
    gemm_wmma_k<3, true, true, false><<<dim3(PE / 64, 4), 128, 0, stream>>>(
        F(3), F(4), F(5), F(6), ein3, nullptr, bigA, 64, PE);
    // et2: 64->16 logits
    gemm_wmma_k<64, false, false, false><<<dim3(PE / 64, 1), 128, 0, stream>>>(
        F(7), F(8), nullptr, nullptr, bigA, nullptr, logit, 16, PE);
    k_softmax16<<<PE / 256, 256, 0, stream>>>(logit, etype);

    // ---- mp1 (nin=2, nout=64, win=1, wout=16) ----
    gemm_wmma_k<2, false, false, false><<<dim3(PN / 64, 64), 128, 0, stream>>>(
        F(9), F(10), nullptr, nullptr, ptsT, nullptr, bigA /*theta*/, 1024, PN);
    gemm_wmma_k<2, false, false, false><<<dim3(PN / 64, 4), 128, 0, stream>>>(
        F(11), F(12), nullptr, nullptr, ptsT, nullptr, selfb, 64, PN);
    k_agg<<<(64 * PN) / 256, 256, 0, stream>>>(etype, bigA, nn, selfb, F(13), F(14), nfA);
    gemm_wmma_k<5, true, true, false><<<dim3(PE / 64, 1), 128, 0, stream>>>(
        F(15), F(16), F(17), F(18), eh5, nullptr, wfA, 16, PE);

    // ---- residual graph-conv block (ri = param base index) ----
    auto res = [&](int ri, const float* nfi, const float* wfi,
                   float* nfo, float* wfo) {
        gemm_wmma_k<64, true, true, false><<<dim3(PN / 64, 4), 128, 0, stream>>>(
            F(ri + 0), F(ri + 1), F(ri + 2), F(ri + 3), nfi, nullptr, n1, 64, PN);
        gemm_wmma_k<16, true, true, false><<<dim3(PE / 64, 1), 128, 0, stream>>>(
            F(ri + 4), F(ri + 5), F(ri + 6), F(ri + 7), wfi, nullptr, w1, 16, PE);
        gemm_wmma_k<64, false, false, false><<<dim3(PN / 64, 64), 128, 0, stream>>>(
            F(ri + 8), F(ri + 9), nullptr, nullptr, n1, nullptr, bigA /*theta*/, 1024, PN);
        gemm_wmma_k<64, false, false, false><<<dim3(PN / 64, 4), 128, 0, stream>>>(
            F(ri + 10), F(ri + 11), nullptr, nullptr, n1, nullptr, selfb, 64, PN);
        k_agg<<<(64 * PN) / 256, 256, 0, stream>>>(etype, bigA, nn, selfb,
                                                   F(ri + 12), F(ri + 13), n1m);
        k_build_eh144<<<dim3(PE / 256, 144), 256, 0, stream>>>(w1, n1, nn, bigA);
        gemm_wmma_k<144, true, true, false><<<dim3(PE / 64, 1), 128, 0, stream>>>(
            F(ri + 14), F(ri + 15), F(ri + 16), F(ri + 17), bigA, nullptr, w1m, 16, PE);
        gemm_wmma_k<64, true, true, true><<<dim3(PN / 64, 4), 128, 0, stream>>>(
            F(ri + 18), F(ri + 19), F(ri + 20), F(ri + 21), n1m, nfi, nfo, 64, PN);
        gemm_wmma_k<16, true, true, true><<<dim3(PE / 64, 1), 128, 0, stream>>>(
            F(ri + 22), F(ri + 23), F(ri + 24), F(ri + 25), w1m, wfi, wfo, 16, PE);
    };
    res(19, nfA, wfA, nfB, wfB);   // res1
    res(45, nfB, wfB, nfA, wfA);   // res2

    // ---- head: global max pool, concat, ctx1 CBR(128->64), ctx2 (64->1) ----
    k_gmax<<<1, 256, 0, stream>>>(nfA, gmx);
    k_build_cat<<<(128 * PN) / 256, 256, 0, stream>>>(nfA, gmx, cat);
    gemm_wmma_k<128, true, true, false><<<dim3(PN / 64, 4), 128, 0, stream>>>(
        F(71), F(72), F(73), F(74), cat, nullptr, hctx, 64, PN);
    k_ctx2<<<PN / 256, 256, 0, stream>>>(F(75), F(76), hctx, (float*)d_out);
}
